// TiteAttention_73057393705645
// MI455X (gfx1250) — matrix operand order repaired and checked by reading, not compiled, not run on hardware
//
#include <hip/hip_runtime.h>
#include <hip/hip_bf16.h>

// ---------------------------------------------------------------------------
// MI455X (gfx1250) TiteAttention: LN -> QKV GEMM (WMMA bf16) -> RoPE ->
// flash attention (WMMA bf16 + async Global->LDS staging) -> O GEMM + residual.
// Matmul FLOPs via v_wmma_f32_16x16x32_bf16; K/V staging via
// global_load_async_to_lds_b128 (ASYNCcnt) double-buffered in LDS.
// GEMM K-loop is unrolled x2 with ping-pong fragment registers so loads for
// step k+1 overlap the 8 WMMAs of step k without register-rotation movs.
// ---------------------------------------------------------------------------

#define TOKENS   16384
#define HIDDEN   768
#define NSEQ     8
#define SEQLEN   2048
#define NHEADS   12
#define HDIM     64

typedef __attribute__((ext_vector_type(16))) __bf16 v16bf;
typedef __attribute__((ext_vector_type(8)))  __bf16 v8bf;
typedef __attribute__((ext_vector_type(8)))  float  v8f;

__device__ __forceinline__ v8f wmma_bf16(v16bf a, v16bf b, v8f c) {
  return __builtin_amdgcn_wmma_f32_16x16x32_bf16(false, a, false, b, (short)0, c, false, false);
}

__device__ __forceinline__ v16bf pack16(v8bf lo, v8bf hi) {
  v16bf r;
#pragma unroll
  for (int i = 0; i < 8; ++i) { r[i] = lo[i]; r[i + 8] = hi[i]; }
  return r;
}

// A operand (16x32 MxK), source row-major [16 rows x stride].
// Lane m=lane&15; half h: element e -> K = (e&7) + ((e>>3)<<4) + h*8.
__device__ __forceinline__ v16bf load_fragA(const __bf16* base, int stride, int k0) {
  const int lane = threadIdx.x & 31;
  const int m = lane & 15, h = lane >> 4;
  const __bf16* p = base + (size_t)m * stride + k0 + h * 8;
  v8bf lo = *reinterpret_cast<const v8bf*>(p);
  v8bf hi = *reinterpret_cast<const v8bf*>(p + 16);
  return pack16(lo, hi);
}

// B operand (32x16 KxN), source stored transposed: [16 N-rows x stride over K].
// Lane n=lane&15; half h: element e -> K = (e&7) + ((e>>3)<<4) + h*8.
__device__ __forceinline__ v16bf load_fragB(const __bf16* base, int stride, int k0) {
  const int lane = threadIdx.x & 31;
  const int n = lane & 15, h = lane >> 4;
  const __bf16* p = base + (size_t)n * stride + k0 + h * 8;
  v8bf lo = *reinterpret_cast<const v8bf*>(p);
  v8bf hi = *reinterpret_cast<const v8bf*>(p + 16);
  return pack16(lo, hi);
}

__device__ __forceinline__ float rowmax16(float v) {
#pragma unroll
  for (int off = 1; off < 16; off <<= 1) v = fmaxf(v, __shfl_xor(v, off, 32));
  return v;
}
__device__ __forceinline__ float rowsum16(float v) {
#pragma unroll
  for (int off = 1; off < 16; off <<= 1) v += __shfl_xor(v, off, 32);
  return v;
}

// Async Global->LDS 16B copy (gfx1250, tracked by ASYNCcnt).
__device__ __forceinline__ void async_copy_b128(uint32_t lds_off, const __bf16* gaddr) {
  asm volatile("global_load_async_to_lds_b128 %0, %1, off"
               :: "v"(lds_off), "v"(gaddr)
               : "memory");
}
__device__ __forceinline__ void wait_async0() {
  asm volatile("s_wait_asynccnt 0x0" ::: "memory");
}

// ---------------------------------------------------------------------------
// 1) LayerNorm: one wave per token -> f32 (residual) + bf16 (GEMM A).
// ---------------------------------------------------------------------------
__global__ __launch_bounds__(256) void ln_kernel(const float* __restrict__ x,
                                                 const float* __restrict__ w,
                                                 const float* __restrict__ b,
                                                 float* __restrict__ xf,
                                                 __bf16* __restrict__ xb) {
  const int token = blockIdx.x * 8 + (threadIdx.x >> 5);
  const int lane = threadIdx.x & 31;
  const float* row = x + (size_t)token * HIDDEN;
  float s = 0.f, ss = 0.f;
  for (int i = lane; i < HIDDEN; i += 32) { float v = row[i]; s += v; ss += v * v; }
#pragma unroll
  for (int off = 1; off < 32; off <<= 1) { s += __shfl_xor(s, off, 32); ss += __shfl_xor(ss, off, 32); }
  const float mean = s * (1.0f / HIDDEN);
  const float var  = ss * (1.0f / HIDDEN) - mean * mean;
  const float rstd = rsqrtf(var + 1e-12f);
  for (int i = lane; i < HIDDEN; i += 32) {
    float y = (row[i] - mean) * rstd * w[i] + b[i];
    xf[(size_t)token * HIDDEN + i] = y;
    xb[(size_t)token * HIDDEN + i] = (__bf16)y;
  }
}

// ---------------------------------------------------------------------------
// 2) Weight transpose + f32->bf16: Wt[n][k] = W[k][n].
// ---------------------------------------------------------------------------
__global__ __launch_bounds__(256) void wt_kernel(const float* __restrict__ W,
                                                 __bf16* __restrict__ Wt) {
  const int i = blockIdx.x * blockDim.x + threadIdx.x;
  if (i >= HIDDEN * HIDDEN) return;
  const int n = i % HIDDEN, k = i / HIDDEN;
  Wt[(size_t)n * HIDDEN + k] = (__bf16)W[i];
}

// ---------------------------------------------------------------------------
// 3/6) WMMA GEMM: Y = A_bf16 @ Wt^T + bias (+residual). Wave computes a
// 32x64 tile (8 WMMA / 32-K step). K-loop unrolled x2 with ping-pong
// fragment registers: refills are issued right after their consumers, so
// they overlap the other phase's 8 WMMAs with no register rotation.
// ---------------------------------------------------------------------------
__global__ __launch_bounds__(256) void gemm_kernel(const __bf16* __restrict__ A,
                                                   const __bf16* __restrict__ Bt,
                                                   const float* __restrict__ bias,
                                                   __bf16* __restrict__ outBf,
                                                   float* __restrict__ outF,
                                                   const float* __restrict__ res) {
  const int wid = blockIdx.x * (blockDim.x >> 5) + (threadIdx.x >> 5);
  const int NT = HIDDEN / 64;                     // 12 column strips
  const int mt = wid / NT, nt = wid % NT;
  if (mt >= TOKENS / 32) return;
  const int mbase = mt * 32, nbase = nt * 64;
  const int lane = threadIdx.x & 31, col = lane & 15, h = lane >> 4;

  const __bf16* A0 = A + (size_t)mbase * HIDDEN;
  const __bf16* A1 = A0 + (size_t)16 * HIDDEN;
  const __bf16* BtN[4];
#pragma unroll
  for (int t = 0; t < 4; ++t) BtN[t] = Bt + (size_t)(nbase + t * 16) * HIDDEN;

  // Ping-pong fragment buffers: X holds k0, Y holds k0+32.
  v16bf a0X = load_fragA(A0, HIDDEN, 0);
  v16bf a1X = load_fragA(A1, HIDDEN, 0);
  v16bf bX[4], bY[4];
#pragma unroll
  for (int t = 0; t < 4; ++t) bX[t] = load_fragB(BtN[t], HIDDEN, 0);
  v16bf a0Y = load_fragA(A0, HIDDEN, 32);
  v16bf a1Y = load_fragA(A1, HIDDEN, 32);
#pragma unroll
  for (int t = 0; t < 4; ++t) bY[t] = load_fragB(BtN[t], HIDDEN, 32);

  v8f acc0[4] = {}, acc1[4] = {};
  for (int k0 = 0; k0 < HIDDEN; k0 += 64) {
    // Phase X: consume k0, refill for k0+64 (overlaps phase Y WMMAs).
#pragma unroll
    for (int t = 0; t < 4; ++t) acc0[t] = wmma_bf16(a0X, bX[t], acc0[t]);
#pragma unroll
    for (int t = 0; t < 4; ++t) acc1[t] = wmma_bf16(a1X, bX[t], acc1[t]);
    if (k0 + 64 < HIDDEN) {
      __builtin_prefetch(A0 + k0 + 128, 0, 3);
      a0X = load_fragA(A0, HIDDEN, k0 + 64);
      a1X = load_fragA(A1, HIDDEN, k0 + 64);
#pragma unroll
      for (int t = 0; t < 4; ++t) bX[t] = load_fragB(BtN[t], HIDDEN, k0 + 64);
    }
    // Phase Y: consume k0+32, refill for k0+96.
#pragma unroll
    for (int t = 0; t < 4; ++t) acc0[t] = wmma_bf16(a0Y, bY[t], acc0[t]);
#pragma unroll
    for (int t = 0; t < 4; ++t) acc1[t] = wmma_bf16(a1Y, bY[t], acc1[t]);
    if (k0 + 96 < HIDDEN) {
      a0Y = load_fragA(A0, HIDDEN, k0 + 96);
      a1Y = load_fragA(A1, HIDDEN, k0 + 96);
#pragma unroll
      for (int t = 0; t < 4; ++t) bY[t] = load_fragB(BtN[t], HIDDEN, k0 + 96);
    }
  }

#pragma unroll
  for (int t = 0; t < 4; ++t) {
#pragma unroll
    for (int r = 0; r < 8; ++r) {
      const int row = r + 8 * h;
      const int n = nbase + t * 16 + col;
      const size_t o0 = (size_t)(mbase + row) * HIDDEN + n;
      const size_t o1 = (size_t)(mbase + 16 + row) * HIDDEN + n;
      const float v0 = acc0[t][r] + bias[n];
      const float v1 = acc1[t][r] + bias[n];
      if (outBf) { outBf[o0] = (__bf16)v0; outBf[o1] = (__bf16)v1; }
      else       { outF[o0] = v0 + res[o0]; outF[o1] = v1 + res[o1]; }
    }
  }
}

// ---------------------------------------------------------------------------
// 4) RoPE + repack: q/k -> [seq*head][pos][d] (rotated), v -> [seq*head][d][pos].
// ---------------------------------------------------------------------------
__global__ __launch_bounds__(256) void rope_pack_kernel(const __bf16* __restrict__ q,
                                                        const __bf16* __restrict__ k,
                                                        const __bf16* __restrict__ v,
                                                        __bf16* __restrict__ qh,
                                                        __bf16* __restrict__ kh,
                                                        __bf16* __restrict__ vth) {
  const int idx = blockIdx.x * blockDim.x + threadIdx.x;
  if (idx >= TOKENS * NHEADS) return;
  const int head = idx % NHEADS;
  const int token = idx / NHEADS;
  const int seq = token / SEQLEN, pos = token % SEQLEN;
  const __bf16* qrow = q + (size_t)token * HIDDEN + head * HDIM;
  const __bf16* krow = k + (size_t)token * HIDDEN + head * HDIM;
  const __bf16* vrow = v + (size_t)token * HIDDEN + head * HDIM;
  const size_t qkBase = ((size_t)(seq * NHEADS + head) * SEQLEN + pos) * HDIM;
  const size_t vBase  = ((size_t)(seq * NHEADS + head) * HDIM) * SEQLEN + pos;
  const float fp = (float)pos;
#pragma unroll 4
  for (int j = 0; j < HDIM / 2; ++j) {
    const float inv = __expf(-(float)j * 0.28782313662425572f);  // ln(10000)/32
    float s, c;
    __sincosf(fp * inv, &s, &c);
    const float q1 = (float)qrow[j], q2 = (float)qrow[j + 32];
    qh[qkBase + j]      = (__bf16)(q1 * c - q2 * s);
    qh[qkBase + j + 32] = (__bf16)(q2 * c + q1 * s);
    const float k1 = (float)krow[j], k2 = (float)krow[j + 32];
    kh[qkBase + j]      = (__bf16)(k1 * c - k2 * s);
    kh[qkBase + j + 32] = (__bf16)(k2 * c + k1 * s);
    vth[vBase + (size_t)j * SEQLEN]        = vrow[j];
    vth[vBase + (size_t)(j + 32) * SEQLEN] = vrow[j + 32];
  }
}

// ---------------------------------------------------------------------------
// 5) Flash attention. Block = 8 waves of the SAME head (8 q-tiles); K/V
// tiles (32 keys) are staged once per block into LDS with async
// global->LDS copies, double-buffered; waves consume B fragments from LDS.
// Per iteration: 4 WMMA for S, online softmax, P via per-wave LDS, 4 WMMA PV.
// Key loop unrolled x2 so the ping-pong buffer index constant-folds.
// ---------------------------------------------------------------------------
__global__ __launch_bounds__(256) void attn_kernel(const __bf16* __restrict__ qh,
                                                   const __bf16* __restrict__ kh,
                                                   const __bf16* __restrict__ vth,
                                                   __bf16* __restrict__ attn) {
  __shared__ __align__(16) __bf16 kbuf[2][32 * HDIM];   // [key][d]
  __shared__ __align__(16) __bf16 vbuf[2][HDIM * 32];   // [d][key]
  __shared__ __align__(16) __bf16 plds[8][16 * 32];     // per-wave P tile

  const int widb = threadIdx.x >> 5;
  const int hb = blockIdx.x >> 4;                 // seq*NHEADS + head
  const int qtile = (blockIdx.x & 15) * 8 + widb;
  const int seq = hb / NHEADS, head = hb % NHEADS;
  const int lane = threadIdx.x & 31, col = lane & 15, h = lane >> 4;
  const int tid = threadIdx.x;

  const size_t hOff = (size_t)hb * SEQLEN * HDIM;
  const __bf16* qp = qh + hOff + (size_t)qtile * 16 * HDIM;
  const __bf16* kp = kh + hOff;                   // [pos][d]
  const __bf16* vp = vth + hOff;                  // [d][pos]

  // Per-thread async-fill slots: K tile 32x64 (256 x 16B), V tile 64x32.
  const int kKey = tid >> 3, kC = tid & 7;        // K: key row, 16B chunk
  const int vD = tid >> 2, vC = tid & 3;          // V: dim row, 16B chunk
  const uint32_t kLds[2] = {
      (uint32_t)(uintptr_t)(&kbuf[0][kKey * HDIM + kC * 8]),
      (uint32_t)(uintptr_t)(&kbuf[1][kKey * HDIM + kC * 8])};
  const uint32_t vLds[2] = {
      (uint32_t)(uintptr_t)(&vbuf[0][vD * 32 + vC * 8]),
      (uint32_t)(uintptr_t)(&vbuf[1][vD * 32 + vC * 8])};

  const v16bf qa0 = load_fragA(qp, HDIM, 0);
  const v16bf qa1 = load_fragA(qp, HDIM, 32);

  float mstate[8], lstate[8];
#pragma unroll
  for (int r = 0; r < 8; ++r) { mstate[r] = -1e30f; lstate[r] = 0.f; }
  v8f O[4] = {};
  __bf16* myp = &plds[widb][0];
  const float scale = 0.125f;  // 1/sqrt(64)

  // Prime buffer 0 with keys [0,32).
  async_copy_b128(kLds[0], kp + (size_t)kKey * HDIM + kC * 8);
  async_copy_b128(vLds[0], vp + (size_t)vD * SEQLEN + vC * 8);

#pragma unroll 2
  for (int kb = 0; kb < SEQLEN; kb += 32) {
    const int cur = (kb >> 5) & 1;
    wait_async0();                 // own fills of buf[cur] done
    __syncthreads();               // all waves' fills visible; buf[cur^1] free
    if (kb + 32 < SEQLEN) {        // fill other buffer for next iteration
      const size_t gk = (size_t)(kb + 32);
      async_copy_b128(kLds[cur ^ 1], kp + (gk + kKey) * HDIM + kC * 8);
      async_copy_b128(vLds[cur ^ 1], vp + (size_t)vD * SEQLEN + gk + vC * 8);
    }

    const __bf16* kt = &kbuf[cur][0];            // [32 keys][64]
    const __bf16* vt = &vbuf[cur][0];            // [64 d][32 keys]

    // S = Q @ K^T : two 16x16 f32 tiles over 32 keys.
    v8f S0 = {}, S1 = {};
    {
      v16bf b0 = load_fragB(kt, HDIM, 0);
      v16bf b1 = load_fragB(kt, HDIM, 32);
      S0 = wmma_bf16(qa0, b0, S0);
      S0 = wmma_bf16(qa1, b1, S0);
      v16bf c0 = load_fragB(kt + 16 * HDIM, HDIM, 0);
      v16bf c1 = load_fragB(kt + 16 * HDIM, HDIM, 32);
      S1 = wmma_bf16(qa0, c0, S1);
      S1 = wmma_bf16(qa1, c1, S1);
    }

    // Online softmax (rows r+8*h, reductions across the 16-lane group).
    float corr[8];
#pragma unroll
    for (int r = 0; r < 8; ++r) {
      const float s0 = S0[r] * scale, s1 = S1[r] * scale;
      const float mx = rowmax16(fmaxf(s0, s1));
      const float nm = fmaxf(mstate[r], mx);
      const float p0 = __expf(s0 - nm), p1 = __expf(s1 - nm);
      myp[(r + 8 * h) * 32 + col]      = (__bf16)p0;
      myp[(r + 8 * h) * 32 + 16 + col] = (__bf16)p1;
      const float sum = rowsum16(p0 + p1);
      corr[r] = __expf(mstate[r] - nm);
      lstate[r] = lstate[r] * corr[r] + sum;
      mstate[r] = nm;
    }
#pragma unroll
    for (int t = 0; t < 4; ++t)
#pragma unroll
      for (int r = 0; r < 8; ++r) O[t][r] *= corr[r];

    // O += P @ V (contraction over 32 keys).
    const v16bf pa = load_fragA(myp, 32, 0);
#pragma unroll
    for (int t = 0; t < 4; ++t) {
      v16bf vb = load_fragB(vt + (size_t)(t * 16) * 32, 32, 0);
      O[t] = wmma_bf16(pa, vb, O[t]);
    }
  }

  // Normalize and scatter to [token][HIDDEN] bf16 for the output GEMM.
#pragma unroll
  for (int t = 0; t < 4; ++t)
#pragma unroll
    for (int r = 0; r < 8; ++r) {
      const int row = r + 8 * h;
      const int token = seq * SEQLEN + qtile * 16 + row;
      const float val = O[t][r] / lstate[r];
      attn[(size_t)token * HIDDEN + head * HDIM + t * 16 + col] = (__bf16)val;
    }
}

// ---------------------------------------------------------------------------
extern "C" void kernel_launch(void* const* d_in, const int* in_sizes, int n_in,
                              void* d_out, int out_size, void* d_ws, size_t ws_size,
                              hipStream_t stream) {
  const float* hs     = (const float*)d_in[0];
  const float* norm_w = (const float*)d_in[3];
  const float* norm_b = (const float*)d_in[4];
  const float* Wq = (const float*)d_in[5];  const float* bq = (const float*)d_in[6];
  const float* Wk = (const float*)d_in[7];  const float* bk = (const float*)d_in[8];
  const float* Wv = (const float*)d_in[9];  const float* bv = (const float*)d_in[10];
  const float* Wo = (const float*)d_in[11]; const float* bo = (const float*)d_in[12];

  char* ws = (char*)d_ws;
  size_t off = 0;
  auto alloc = [&](size_t bytes) -> char* {
    char* p = ws + off;
    off += (bytes + 255) & ~(size_t)255;
    return p;
  };
  const size_t actF = (size_t)TOKENS * HIDDEN * sizeof(float);
  const size_t actB = (size_t)TOKENS * HIDDEN * sizeof(__bf16);
  const size_t wB   = (size_t)HIDDEN * HIDDEN * sizeof(__bf16);

  float*  xf  = (float*) alloc(actF);    // LN output f32 (residual)
  __bf16* xb  = (__bf16*)alloc(actB);    // LN output bf16 (GEMM A)
  __bf16* WqT = (__bf16*)alloc(wB);
  __bf16* WkT = (__bf16*)alloc(wB);
  __bf16* WvT = (__bf16*)alloc(wB);
  __bf16* WoT = (__bf16*)alloc(wB);
  __bf16* qt  = (__bf16*)alloc(actB);    // raw projections [token][hidden]
  __bf16* kt  = (__bf16*)alloc(actB);
  __bf16* vt  = (__bf16*)alloc(actB);
  __bf16* qhd = (__bf16*)alloc(actB);    // [seq*head][pos][d], rope'd
  __bf16* khd = (__bf16*)alloc(actB);
  __bf16* vtd = (__bf16*)alloc(actB);    // [seq*head][d][pos]
  __bf16* atn = qt;                      // qt dead after rope: alias

  const dim3 blk(256);
  const int GEMM_BLOCKS = (TOKENS / 32) * (HIDDEN / 64) / 8;     // 768
  const int ATTN_BLOCKS = NSEQ * NHEADS * (SEQLEN / 16) / 8;     // 1536

  ln_kernel<<<TOKENS / 8, blk, 0, stream>>>(hs, norm_w, norm_b, xf, xb);

  wt_kernel<<<(HIDDEN * HIDDEN + 255) / 256, blk, 0, stream>>>(Wq, WqT);
  wt_kernel<<<(HIDDEN * HIDDEN + 255) / 256, blk, 0, stream>>>(Wk, WkT);
  wt_kernel<<<(HIDDEN * HIDDEN + 255) / 256, blk, 0, stream>>>(Wv, WvT);
  wt_kernel<<<(HIDDEN * HIDDEN + 255) / 256, blk, 0, stream>>>(Wo, WoT);

  gemm_kernel<<<GEMM_BLOCKS, blk, 0, stream>>>(xb, WqT, bq, qt, nullptr, nullptr);
  gemm_kernel<<<GEMM_BLOCKS, blk, 0, stream>>>(xb, WkT, bk, kt, nullptr, nullptr);
  gemm_kernel<<<GEMM_BLOCKS, blk, 0, stream>>>(xb, WvT, bv, vt, nullptr, nullptr);

  rope_pack_kernel<<<(TOKENS * NHEADS + 255) / 256, blk, 0, stream>>>(qt, kt, vt, qhd, khd, vtd);

  attn_kernel<<<ATTN_BLOCKS, blk, 0, stream>>>(qhd, khd, vtd, atn);

  gemm_kernel<<<GEMM_BLOCKS, blk, 0, stream>>>(atn, WoT, bo, nullptr, (float*)d_out, xf);
}
